// MultiDirectionalSpatialScanner_14688788152946
// MI455X (gfx1250) — matrix operand order repaired and checked by reading, not compiled, not run on hardware
//
#include <hip/hip_runtime.h>

typedef __attribute__((ext_vector_type(16))) _Float16 v16h;
typedef __attribute__((ext_vector_type(8)))  _Float16 v8h;
typedef __attribute__((ext_vector_type(8)))  float    v8f;
typedef unsigned int u32;
typedef __attribute__((ext_vector_type(4))) u32 u32x4;
typedef __attribute__((ext_vector_type(8))) u32 u32x8;

static constexpr int Bsz  = 16;
static constexpr int Hh   = 24;
static constexpr int Ww   = 24;
static constexpr int Ntok = Hh * Ww;        // 576
static constexpr int Dm   = 768;
static constexpr int DI   = 1536;           // d_inner
static constexpr int DS   = 16;             // d_state
static constexpr int RK   = 48;             // dt_rank
static constexpr int DBLW = RK + 2 * DS;    // 80 (real dbl width)
static constexpr int DBLP = 128;            // padded dbl width (aligned GEMM)
static constexpr int RKP  = 64;             // padded dt K (zero weight rows 48..63)
static constexpr int NDIR = 4;
static constexpr long BN  = (long)Bsz * Ntok; // 9216

__device__ __forceinline__ float silu_f(float x) { return x / (1.f + __expf(-x)); }
__device__ __forceinline__ float softplus_f(float x) { return (x > 20.f) ? x : log1pf(__expf(x)); }

// ---------------------------------------------------------------------------
// Tensor Data Mover: async 2D f16 tile (tile0 x tile1 elements) global -> LDS.
// D# pad feature: 4 DWORDs after every 16 DWORDs => LDS row stride 40 halfs
// (bank-conflict-friendly), matching the WMMA fragment loaders below.
// Descriptor per CDNA5 ISA ch.8, groups 0/1 only (VADDR2/3 NULL => 2D tensor).
// ---------------------------------------------------------------------------
__device__ __forceinline__ void tdm_load_tile_f16(const void* gaddr, unsigned lds_off,
                                                  u32 tile0, u32 tile1, u32 stride0) {
    unsigned long long ga = (unsigned long long)gaddr;
    u32x4 g0;
    g0[0] = 1u;                                   // count=1, user descriptor
    g0[1] = lds_off;                              // LDS byte address
    g0[2] = (u32)ga;                              // global_addr[31:0]
    g0[3] = ((u32)(ga >> 32) & 0x01FFFFFFu)       // global_addr[56:32]
          | (2u << 30);                           // type = 2 ("image")
    u32x8 g1;
    g1[0] = (1u << 16)                            // data_size = 2 bytes
          | (1u << 20)                            // pad_enable
          | (3u << 22)                            // pad_interval: 16 DWORDs
          | (3u << 25);                           // pad_amount : 4 DWORDs
    u32 tdim0 = stride0;                          // tensor dim0 (OOB) = row length
    u32 tdim1 = 1u << 20;                         // aligned path: no OOB clipping
    g1[1] = (tdim0 & 0xFFFFu) << 16;              // tensor_dim0[15:0]
    g1[2] = (tdim0 >> 16) | ((tdim1 & 0xFFFFu) << 16);
    g1[3] = (tdim1 >> 16) | (tile0 << 16);        // tile_dim0
    g1[4] = tile1;                                // tile_dim1 (tile_dim2 = 0)
    g1[5] = stride0;                              // tensor_dim0_stride[31:0]
    g1[6] = 0u;
    g1[7] = 0u;
    asm volatile("tensor_load_to_lds %0, %1" :: "s"(g0), "s"(g1) : "memory");
}

// ---------------------------------------------------------------------------
// Diagonal permutation inverse (inv_perm[token] = position in diagonal scan)
// ---------------------------------------------------------------------------
__global__ void mdss_build_invp(int* __restrict__ invp) {
    __shared__ int perm[Ntok];
    if (threadIdx.x == 0) {
        int c = 0;
        for (int off = -(Hh - 1); off < Ww; ++off)
            for (int i = 0; i < Hh; ++i) {
                int j = i + off;
                if (0 <= j && j < Ww) perm[c++] = i * Ww + j;
            }
    }
    __syncthreads();
    for (int t = threadIdx.x; t < Ntok; t += blockDim.x) invp[perm[t]] = t;
}

// ---------------------------------------------------------------------------
// Weight convert with zero-padding: f32 (K x N) row-major -> f16 (Np x Kp),
// dst[n][k] = (k<K && n<N) ? src[k][n] : 0.  Makes every GEMM shape aligned.
// ---------------------------------------------------------------------------
__global__ __launch_bounds__(256)
void mdss_wconvert(const float* __restrict__ src, _Float16* __restrict__ dst,
                   int K, int Nn, int Kp, int Np) {
    long idx = (long)blockIdx.x * 256 + threadIdx.x;
    if (idx >= (long)Kp * Np) return;
    int n = (int)(idx / Kp);
    int k = (int)(idx - (long)n * Kp);
    dst[idx] = (k < K && n < Nn) ? (_Float16)src[(long)k * Nn + n] : (_Float16)0.f;
}

// ---------------------------------------------------------------------------
// Input LayerNorm + sinusoidal pos-embed, emit 4 permuted f16 scan orders
// ---------------------------------------------------------------------------
__global__ __launch_bounds__(256)
void mdss_ln_pos(const float* __restrict__ x, const float* __restrict__ g,
                 const float* __restrict__ bb, const int* __restrict__ invp,
                 _Float16* __restrict__ xperm) {
    const int token = blockIdx.x, b = blockIdx.y;
    const float* row = x + ((long)b * Ntok + token) * Dm;
    __shared__ float r0[8], r1[8];
    float s = 0.f, s2 = 0.f;
    for (int c = threadIdx.x; c < Dm; c += 256) { float v = row[c]; s += v; s2 += v * v; }
    for (int o = 16; o > 0; o >>= 1) { s += __shfl_xor(s, o, 32); s2 += __shfl_xor(s2, o, 32); }
    if ((threadIdx.x & 31) == 0) { r0[threadIdx.x >> 5] = s; r1[threadIdx.x >> 5] = s2; }
    __syncthreads();
    float S = 0.f, S2 = 0.f;
    for (int i = 0; i < 8; ++i) { S += r0[i]; S2 += r1[i]; }
    const float mean = S / (float)Dm;
    const float var  = S2 / (float)Dm - mean * mean;
    const float rstd = rsqrtf(var + 1e-5f);

    const int hh = token / Ww, ww = token % Ww;
    const int s1 = Ntok - 1 - token;
    const int s2i = ww * Hh + hh;
    const int s3 = invp[token];
    const float ph = (float)hh / (float)(Hh - 1) * 2.f - 1.f;
    const float pw = (float)ww / (float)(Ww - 1) * 2.f - 1.f;
    const long bb0 = (long)b * Ntok;

    for (int c = threadIdx.x; c < Dm; c += 256) {
        float v = (row[c] - mean) * rstd * g[c] + bb[c];
        int q = c >> 2, r = c & 3;
        float div = __expf(-(float)(4 * q) * (9.2103403719761836f / (float)Dm));
        float pv = (r == 0) ? __sinf(ph * div) : (r == 1) ? __cosf(ph * div)
                 : (r == 2) ? __sinf(pw * div) : __cosf(pw * div);
        _Float16 hv = (_Float16)(v + pv);
        xperm[0L * BN * Dm + (bb0 + token) * Dm + c] = hv;
        xperm[1L * BN * Dm + (bb0 + s1)    * Dm + c] = hv;
        xperm[2L * BN * Dm + (bb0 + s2i)   * Dm + c] = hv;
        xperm[3L * BN * Dm + (bb0 + s3)    * Dm + c] = hv;
    }
}

// ---------------------------------------------------------------------------
// WMMA fragment helpers (ISA 7.12.2 layouts), LDS row stride = 40 halfs
// ---------------------------------------------------------------------------
__device__ __forceinline__ v16h frag_a(const _Float16 (*As)[40], int rb, int halfsel) {
    v8h lo = *(const v8h*)&As[rb][halfsel * 8];
    v8h hi = *(const v8h*)&As[rb][16 + halfsel * 8];
    return __builtin_shufflevector(lo, hi, 0, 1, 2, 3, 4, 5, 6, 7,
                                   8, 9, 10, 11, 12, 13, 14, 15);
}
__device__ __forceinline__ v16h frag_b(const _Float16 (*Bs)[40], int nb, int halfsel) {
    v8h lo = *(const v8h*)&Bs[nb][halfsel * 8];
    v8h hi = *(const v8h*)&Bs[nb][16 + halfsel * 8];
    return __builtin_shufflevector(lo, hi, 0, 1, 2, 3, 4, 5, 6, 7,
                                   8, 9, 10, 11, 12, 13, 14, 15);
}

// ---------------------------------------------------------------------------
// Aligned WMMA GEMM, double-buffered TDM staging. M%128==0, N%128==0, K%32==0.
// C(MxN) = A(MxK, stride lda) * Bt(NxK). Block 128x128x32, 8 waves, 8 WMMA/wave.
// Wave 0 drives the Tensor Data Mover: issues the k+1 tile pair before waiting
// on the k tile pair (s_wait_tensorcnt 2 -> older pair landed; TDM ops from one
// wave complete in order). End-of-step barrier guarantees a buffer is fully
// consumed before the DMA that overwrites it is issued next iteration.
// ---------------------------------------------------------------------------
template <bool OUT_F16, bool SILU>
__global__ __launch_bounds__(256)
void mdss_gemm_wmma_tdm(const _Float16* __restrict__ A, const _Float16* __restrict__ Bt,
                        void* __restrict__ Cv, int M, int Nn, int K, int lda) {
    __shared__ __align__(16) _Float16 As[2][128][40];
    __shared__ __align__(16) _Float16 Bs[2][128][40];
    const int tid = threadIdx.x;
    const int lane = tid & 31, wv = tid >> 5;
    const int wm = wv >> 1, wn = wv & 1;
    const int m0 = blockIdx.y * 128, n0 = blockIdx.x * 128;
    const int halfsel = lane >> 4, l15 = lane & 15;

    v8f acc[2][4];
    {
        v8f z = {};
        for (int i = 0; i < 2; ++i)
            for (int j = 0; j < 4; ++j) acc[i][j] = z;
    }

    if (wv == 0) {  // prologue: DMA pair for buffer 0
        tdm_load_tile_f16(A + (long)m0 * lda, (unsigned)(uintptr_t)&As[0][0][0], 32u, 128u, (u32)lda);
        tdm_load_tile_f16(Bt + (long)n0 * K, (unsigned)(uintptr_t)&Bs[0][0][0], 32u, 128u, (u32)K);
    }

    int p = 0;
    for (int k0 = 0; k0 < K; k0 += 32, p ^= 1) {
        if (wv == 0) {
            if (k0 + 32 < K) {
                tdm_load_tile_f16(A + (long)m0 * lda + k0 + 32,
                                  (unsigned)(uintptr_t)&As[p ^ 1][0][0], 32u, 128u, (u32)lda);
                tdm_load_tile_f16(Bt + (long)n0 * K + k0 + 32,
                                  (unsigned)(uintptr_t)&Bs[p ^ 1][0][0], 32u, 128u, (u32)K);
                __builtin_amdgcn_s_wait_tensorcnt(2);  // buffer-p pair complete
            } else {
                __builtin_amdgcn_s_wait_tensorcnt(0);
            }
        }
        __syncthreads();

        v16h af[2], bf[4];
#pragma unroll
        for (int fi = 0; fi < 2; ++fi) af[fi] = frag_a(As[p], wm * 32 + fi * 16 + l15, halfsel);
#pragma unroll
        for (int fj = 0; fj < 4; ++fj) bf[fj] = frag_b(Bs[p], wn * 64 + fj * 16 + l15, halfsel);
#pragma unroll
        for (int fi = 0; fi < 2; ++fi)
#pragma unroll
            for (int fj = 0; fj < 4; ++fj)
                acc[fi][fj] = __builtin_amdgcn_wmma_f32_16x16x32_f16(
                    false, af[fi], false, bf[fj], (short)0, acc[fi][fj], false, false);
        __syncthreads();  // reads of buffer p done before it is re-filled
    }

#pragma unroll
    for (int fi = 0; fi < 2; ++fi)
#pragma unroll
        for (int fj = 0; fj < 4; ++fj)
#pragma unroll
            for (int r = 0; r < 8; ++r) {
                int row = m0 + wm * 32 + fi * 16 + halfsel * 8 + r;
                int col = n0 + wn * 64 + fj * 16 + l15;
                float v = acc[fi][fj][r];
                if (SILU) v = silu_f(v);
                if (OUT_F16) ((_Float16*)Cv)[(long)row * Nn + col] = (_Float16)v;
                else ((float*)Cv)[(long)row * Nn + col] = v;
            }
}

// ---------------------------------------------------------------------------
// Causal depthwise conv (D_CONV=4) + SiLU on xi = xz[:, :DI]
// ---------------------------------------------------------------------------
__global__ __launch_bounds__(256)
void mdss_conv_silu(const _Float16* __restrict__ xz, const float* __restrict__ cw,
                    _Float16* __restrict__ xc) {
    long idx = (long)blockIdx.x * 256 + threadIdx.x;
    if (idx >= BN * (long)DI) return;
    int ch = (int)(idx % DI);
    long bt = idx / DI;
    int t = (int)(bt % Ntok);
    long b = bt / Ntok;
    float a = 0.f;
#pragma unroll
    for (int k = 0; k < 4; ++k) {
        int tt = t + k - 3;
        if (tt >= 0)
            a += cw[ch * 4 + k] * (float)xz[((long)b * Ntok + tt) * (2 * DI) + ch];
    }
    xc[idx] = (_Float16)silu_f(a);
}

// ---------------------------------------------------------------------------
// Selective scan: one lane per (b, channel), 16-float state in registers.
// dbl has padded row stride DBLP.
// ---------------------------------------------------------------------------
__global__ __launch_bounds__(256)
void mdss_scan(const _Float16* __restrict__ dbl, const _Float16* __restrict__ dtraw,
               const _Float16* __restrict__ xc, const _Float16* __restrict__ xz,
               const float* __restrict__ A_log, const float* __restrict__ Dp,
               const float* __restrict__ dtb, _Float16* __restrict__ yact) {
    const int b = blockIdx.y;
    const int ch = blockIdx.x * 256 + threadIdx.x;
    float An[DS], h[DS];
#pragma unroll
    for (int n = 0; n < DS; ++n) { An[n] = -__expf(A_log[(long)ch * DS + n]); h[n] = 0.f; }
    const float Dv = Dp[ch];
    const float bias = dtb[ch];
    __shared__ _Float16 Bsm[2][DS], Csm[2][DS];

    for (int t = 0; t < Ntok; ++t) {
        const long idx = (long)b * Ntok + t;
        const int buf = t & 1;
        if (threadIdx.x < DS)
            Bsm[buf][threadIdx.x] = dbl[idx * DBLP + RK + threadIdx.x];
        else if (threadIdx.x < 2 * DS)
            Csm[buf][threadIdx.x - DS] = dbl[idx * DBLP + RK + DS + (threadIdx.x - DS)];
        __syncthreads();

        float dtv = softplus_f((float)dtraw[idx * DI + ch] + bias);
        float xt  = (float)xc[idx * DI + ch];
        float y = 0.f;
        const float xdt = xt * dtv;
#pragma unroll
        for (int n = 0; n < DS; ++n) {
            float e = __expf(An[n] * dtv);
            h[n] = h[n] * e + (float)Bsm[buf][n] * xdt;
            y += h[n] * (float)Csm[buf][n];
        }
        y += Dv * xt;
        float z = (float)xz[idx * (2 * DI) + DI + ch];
        y *= silu_f(z);
        yact[idx * DI + ch] = (_Float16)y;
    }
}

// ---------------------------------------------------------------------------
// Un-permute direction output; append to concat; accumulate weighted fuse.
// ---------------------------------------------------------------------------
__global__ __launch_bounds__(256)
void mdss_unperm(const _Float16* __restrict__ ypd, _Float16* __restrict__ concat,
                 float* __restrict__ fusedacc, const float* __restrict__ dw,
                 const int* __restrict__ invp, int d) {
    long idx = (long)blockIdx.x * 256 + threadIdx.x;
    if (idx >= BN * (long)Dm) return;
    int c = (int)(idx % Dm);
    long bt = idx / Dm;
    int t = (int)(bt % Ntok);
    long b = bt / Ntok;
    int s;
    if (d == 0) s = t;
    else if (d == 1) s = Ntok - 1 - t;
    else if (d == 2) { int hh = t / Ww, ww = t % Ww; s = ww * Hh + hh; }
    else s = invp[t];
    float v = (float)ypd[((long)b * Ntok + s) * Dm + c];
    concat[bt * (long)(NDIR * Dm) + d * Dm + c] = (_Float16)v;

    float w0 = dw[0], w1 = dw[1], w2 = dw[2], w3 = dw[3];
    float mx = fmaxf(fmaxf(w0, w1), fmaxf(w2, w3));
    float e0 = __expf(w0 - mx), e1 = __expf(w1 - mx), e2 = __expf(w2 - mx), e3 = __expf(w3 - mx);
    float wd = ((d == 0) ? e0 : (d == 1) ? e1 : (d == 2) ? e2 : e3) / (e0 + e1 + e2 + e3);
    float prev = (d == 0) ? 0.f : fusedacc[idx];
    fusedacc[idx] = prev + wd * v;
}

// ---------------------------------------------------------------------------
// Final residual + LayerNorm
// ---------------------------------------------------------------------------
__global__ __launch_bounds__(256)
void mdss_ln_out(const float* __restrict__ vis, const float* __restrict__ fa,
                 const float* __restrict__ fu, const float* __restrict__ g,
                 const float* __restrict__ bb, float* __restrict__ out) {
    const int token = blockIdx.x, b = blockIdx.y;
    const long base = ((long)b * Ntok + token) * Dm;
    __shared__ float r0[8], r1[8];
    float s = 0.f, s2 = 0.f;
    for (int c = threadIdx.x; c < Dm; c += 256) {
        float v = vis[base + c] + fa[base + c] + fu[base + c];
        s += v; s2 += v * v;
    }
    for (int o = 16; o > 0; o >>= 1) { s += __shfl_xor(s, o, 32); s2 += __shfl_xor(s2, o, 32); }
    if ((threadIdx.x & 31) == 0) { r0[threadIdx.x >> 5] = s; r1[threadIdx.x >> 5] = s2; }
    __syncthreads();
    float S = 0.f, S2 = 0.f;
    for (int i = 0; i < 8; ++i) { S += r0[i]; S2 += r1[i]; }
    const float mean = S / (float)Dm;
    const float rstd = rsqrtf(S2 / (float)Dm - mean * mean + 1e-5f);
    for (int c = threadIdx.x; c < Dm; c += 256) {
        float v = vis[base + c] + fa[base + c] + fu[base + c];
        out[base + c] = (v - mean) * rstd * g[c] + bb[c];
    }
}

// ---------------------------------------------------------------------------
extern "C" void kernel_launch(void* const* d_in, const int* in_sizes, int n_in,
                              void* d_out, int out_size, void* d_ws, size_t ws_size,
                              hipStream_t stream) {
    const float* vision     = (const float*)d_in[0];
    const float* ln_in_g    = (const float*)d_in[3];
    const float* ln_in_b    = (const float*)d_in[4];
    const float* dir_proj_w = (const float*)d_in[5];
    const float* in_proj_w  = (const float*)d_in[6];
    const float* conv_w     = (const float*)d_in[7];
    const float* x_proj_w   = (const float*)d_in[8];
    const float* dt_proj_w  = (const float*)d_in[9];
    const float* dt_proj_b  = (const float*)d_in[10];
    const float* A_log      = (const float*)d_in[11];
    const float* D_param    = (const float*)d_in[12];
    const float* out_proj_w = (const float*)d_in[13];
    const float* fusion_w1  = (const float*)d_in[14];
    const float* fusion_w2  = (const float*)d_in[15];
    const float* ln_out_g   = (const float*)d_in[16];
    const float* ln_out_b   = (const float*)d_in[17];
    const float* dir_wts    = (const float*)d_in[18];
    float* out = (float*)d_out;

    char* ws = (char*)d_ws;
    size_t off = 0;
    auto alloc = [&](size_t bytes) -> char* {
        char* p = ws + off;
        off += (bytes + 255) & ~(size_t)255;
        return p;
    };
    int*      invp   = (int*)alloc(Ntok * sizeof(int));
    _Float16* wdirT  = (_Float16*)alloc((size_t)NDIR * Dm * Dm * 2);
    _Float16* winT   = (_Float16*)alloc((size_t)NDIR * 2 * DI * Dm * 2);
    _Float16* wxpT   = (_Float16*)alloc((size_t)NDIR * DBLP * DI * 2);   // padded N 80->128
    _Float16* wdtT   = (_Float16*)alloc((size_t)NDIR * DI * RKP * 2);    // padded K 48->64
    _Float16* woutT  = (_Float16*)alloc((size_t)NDIR * Dm * DI * 2);
    _Float16* wf1T   = (_Float16*)alloc((size_t)2 * Dm * NDIR * Dm * 2);
    _Float16* wf2T   = (_Float16*)alloc((size_t)Dm * 2 * Dm * 2);
    _Float16* xperm  = (_Float16*)alloc((size_t)NDIR * BN * Dm * 2);
    _Float16* xp     = (_Float16*)alloc((size_t)BN * Dm * 2);
    _Float16* xz     = (_Float16*)alloc((size_t)BN * 2 * DI * 2);
    _Float16* xc     = (_Float16*)alloc((size_t)BN * DI * 2);
    _Float16* dblp   = (_Float16*)alloc((size_t)BN * DBLP * 2);          // padded dbl
    _Float16* dtraw  = (_Float16*)alloc((size_t)BN * DI * 2);
    _Float16* yact   = (_Float16*)alloc((size_t)BN * DI * 2);
    _Float16* ypd    = (_Float16*)alloc((size_t)BN * Dm * 2);
    _Float16* concat = (_Float16*)alloc((size_t)BN * NDIR * Dm * 2);
    float*    fused  = (float*)alloc((size_t)BN * Dm * 4);
    _Float16* hbuf   = (_Float16*)alloc((size_t)BN * DI * 2);
    float*    fusion = (float*)alloc((size_t)BN * Dm * 4);

    mdss_build_invp<<<1, Ntok, 0, stream>>>(invp);

    auto wconv = [&](const float* s, _Float16* dd, int K, int Nn, int Kp, int Np) {
        long tot = (long)Kp * Np;
        mdss_wconvert<<<dim3((unsigned)((tot + 255) / 256)), 256, 0, stream>>>(s, dd, K, Nn, Kp, Np);
    };
    for (int d = 0; d < NDIR; ++d) {
        wconv(dir_proj_w + (size_t)d * Dm * Dm,     wdirT + (size_t)d * Dm * Dm,     Dm, Dm,   Dm, Dm);
        wconv(in_proj_w  + (size_t)d * Dm * 2 * DI, winT  + (size_t)d * 2 * DI * Dm, Dm, 2 * DI, Dm, 2 * DI);
        wconv(x_proj_w   + (size_t)d * DI * DBLW,   wxpT  + (size_t)d * DBLP * DI,   DI, DBLW, DI, DBLP);
        wconv(dt_proj_w  + (size_t)d * RK * DI,     wdtT  + (size_t)d * DI * RKP,    RK, DI,   RKP, DI);
        wconv(out_proj_w + (size_t)d * DI * Dm,     woutT + (size_t)d * Dm * DI,     DI, Dm,   DI, Dm);
    }
    wconv(fusion_w1, wf1T, NDIR * Dm, 2 * Dm, NDIR * Dm, 2 * Dm);
    wconv(fusion_w2, wf2T, 2 * Dm, Dm, 2 * Dm, Dm);

    mdss_ln_pos<<<dim3(Ntok, Bsz), 256, 0, stream>>>(vision, ln_in_g, ln_in_b, invp, xperm);

    auto ggrid = [](long M, int Nn) { return dim3((unsigned)(Nn / 128), (unsigned)(M / 128)); };

    for (int d = 0; d < NDIR; ++d) {
        mdss_gemm_wmma_tdm<true, false><<<ggrid(BN, Dm), 256, 0, stream>>>(
            xperm + (size_t)d * BN * Dm, wdirT + (size_t)d * Dm * Dm, xp, (int)BN, Dm, Dm, Dm);
        mdss_gemm_wmma_tdm<true, false><<<ggrid(BN, 2 * DI), 256, 0, stream>>>(
            xp, winT + (size_t)d * 2 * DI * Dm, xz, (int)BN, 2 * DI, Dm, Dm);
        mdss_conv_silu<<<dim3((unsigned)((BN * DI + 255) / 256)), 256, 0, stream>>>(
            xz, conv_w + (size_t)d * DI * 4, xc);
        mdss_gemm_wmma_tdm<true, false><<<ggrid(BN, DBLP), 256, 0, stream>>>(
            xc, wxpT + (size_t)d * DBLP * DI, dblp, (int)BN, DBLP, DI, DI);
        mdss_gemm_wmma_tdm<true, false><<<ggrid(BN, DI), 256, 0, stream>>>(
            dblp, wdtT + (size_t)d * DI * RKP, dtraw, (int)BN, DI, RKP, DBLP);
        mdss_scan<<<dim3(DI / 256, Bsz), 256, 0, stream>>>(
            dblp, dtraw, xc, xz, A_log + (size_t)d * DI * DS, D_param + (size_t)d * DI,
            dt_proj_b + (size_t)d * DI, yact);
        mdss_gemm_wmma_tdm<true, false><<<ggrid(BN, Dm), 256, 0, stream>>>(
            yact, woutT + (size_t)d * Dm * DI, ypd, (int)BN, Dm, DI, DI);
        mdss_unperm<<<dim3((unsigned)((BN * Dm + 255) / 256)), 256, 0, stream>>>(
            ypd, concat, fused, dir_wts, invp, d);
    }

    mdss_gemm_wmma_tdm<true, true><<<ggrid(BN, 2 * Dm), 256, 0, stream>>>(
        concat, wf1T, hbuf, (int)BN, 2 * Dm, NDIR * Dm, NDIR * Dm);
    mdss_gemm_wmma_tdm<false, false><<<ggrid(BN, Dm), 256, 0, stream>>>(
        hbuf, wf2T, fusion, (int)BN, Dm, 2 * Dm, 2 * Dm);
    mdss_ln_out<<<dim3(Ntok, Bsz), 256, 0, stream>>>(vision, fused, fusion, ln_out_g, ln_out_b, out);
}
